// GATLayer_58136677318720
// MI455X (gfx1250) — compile-verified, matrix operand order checked by reading, not run on hardware
//
#include <hip/hip_runtime.h>
#include <math.h>

#define NN    8192
#define DIN   512
#define DOUT  256
#define LOG2E 1.4426950408889634f
#define NEGINF_S (-9e15f * 1.4426950408889634f)   // NEG_INF in the log2 domain

typedef __attribute__((ext_vector_type(16))) __bf16 v16bf;
typedef __attribute__((ext_vector_type(8)))  float  v8f;

union Frag16 {
    v16bf v;
    unsigned int u[8];
    uint4 q[2];
};

__device__ __forceinline__ unsigned short f32_to_bf16(float x) {
    union { __bf16 h; unsigned short u; } c;
    c.h = (__bf16)x;                       // RNE hardware convert
    return c.u;
}

__device__ __forceinline__ unsigned int pack_bf16x2(float lo, float hi) {
    union { __bf16 h[2]; unsigned int u; } c;
    c.h[0] = (__bf16)lo;
    c.h[1] = (__bf16)hi;                   // should fuse to v_cvt_pk_bf16_f32
    return c.u;
}

// ---------------- conversion kernels ----------------
__global__ __launch_bounds__(256) void cvt_x_kernel(const float* __restrict__ x,
                                                    unsigned short* __restrict__ xbf) {
    size_t i = (size_t)blockIdx.x * 256 + threadIdx.x;
    if (i < (size_t)NN * DIN) xbf[i] = f32_to_bf16(x[i]);
}

__global__ __launch_bounds__(256) void cvt_w_kernel(const float* __restrict__ W,
                                                    unsigned short* __restrict__ WT) {
    size_t i = (size_t)blockIdx.x * 256 + threadIdx.x;
    if (i < (size_t)DIN * DOUT) {
        int k = (int)(i / DOUT);
        int n = (int)(i % DOUT);
        WT[(size_t)n * DIN + k] = f32_to_bf16(W[i]);
    }
}

// ---------------- GEMM1: h = x @ W (bf16 WMMA, f32 acc) ----------------
// grid (NN/16, DOUT/32), block 64 (2 waves); wave w -> 16x16 tile at col (by*2+w)*16
__global__ __launch_bounds__(64) void gemm1_kernel(const unsigned short* __restrict__ xbf,
                                                   const unsigned short* __restrict__ WT,
                                                   float* __restrict__ h,
                                                   unsigned short* __restrict__ hT) {
    const int wave = threadIdx.x >> 5;
    const int lane = threadIdx.x & 31;
    const int m  = lane & 15;
    const int hh = lane >> 4;
    const int row0 = blockIdx.x * 16;
    const int col0 = (blockIdx.y * 2 + wave) * 16;

    const unsigned short* arow = xbf + (size_t)(row0 + m) * DIN;
    const unsigned short* brow = WT  + (size_t)(col0 + m) * DIN;

    v8f c = {0.f, 0.f, 0.f, 0.f, 0.f, 0.f, 0.f, 0.f};
    #pragma unroll 4
    for (int k0 = 0; k0 < DIN; k0 += 32) {
        Frag16 a, b;
        const int kb0 = k0 + 8 * hh;
        const int kb1 = kb0 + 16;
        a.q[0] = *(const uint4*)(arow + kb0);
        a.q[1] = *(const uint4*)(arow + kb1);
        b.q[0] = *(const uint4*)(brow + kb0);
        b.q[1] = *(const uint4*)(brow + kb1);
        c = __builtin_amdgcn_wmma_f32_16x16x32_bf16(false, a.v, false, b.v,
                                                    (short)0, c, false, false);
    }
    #pragma unroll
    for (int r = 0; r < 8; ++r) {
        const int row = row0 + r + 8 * hh;
        const int col = col0 + m;
        const float v = c[r];
        h [(size_t)row * DOUT + col] = v;
        hT[(size_t)col * NN   + row] = f32_to_bf16(v);
    }
}

// ---------------- f1/f2 = (h@a1, h@a2) * log2(e), wave per row ----------------
__global__ __launch_bounds__(256) void f12_kernel(const float* __restrict__ h,
                                                  const float* __restrict__ a1,
                                                  const float* __restrict__ a2,
                                                  float* __restrict__ f1s,
                                                  float* __restrict__ f2s) {
    const int wave = threadIdx.x >> 5;
    const int lane = threadIdx.x & 31;
    const int row  = blockIdx.x * 8 + wave;
    const float* hr = h + (size_t)row * DOUT;
    float s1 = 0.f, s2 = 0.f;
    for (int t = lane; t < DOUT; t += 32) {
        const float v = hr[t];
        s1 += v * a1[t];
        s2 += v * a2[t];
    }
    #pragma unroll
    for (int off = 16; off > 0; off >>= 1) {
        s1 += __shfl_xor(s1, off, 32);
        s2 += __shfl_xor(s2, off, 32);
    }
    if (lane == 0) { f1s[row] = s1 * LOG2E; f2s[row] = s2 * LOG2E; }
}

// ---------------- per-row masked-softmax stats in log2 domain ----------------
__global__ __launch_bounds__(256) void rowstats_kernel(const int* __restrict__ adj,
                                                       const float* __restrict__ f1s,
                                                       const float* __restrict__ f2s,
                                                       float* __restrict__ mxs,
                                                       float* __restrict__ invZ) {
    __shared__ float red[256];
    const int row = blockIdx.x;
    const int* arow = adj + (size_t)row * NN;
    const float f1r = f1s[row];

    float lm = NEGINF_S;
    for (int j = threadIdx.x; j < NN; j += 256) {
        float e = f1r + f2s[j];
        e = fmaxf(e, 0.01f * e);                     // leaky_relu (scaled domain)
        const float s = (arow[j] > 0) ? e : NEGINF_S;
        lm = fmaxf(lm, s);
    }
    red[threadIdx.x] = lm; __syncthreads();
    for (int s = 128; s > 0; s >>= 1) {
        if (threadIdx.x < s) red[threadIdx.x] = fmaxf(red[threadIdx.x], red[threadIdx.x + s]);
        __syncthreads();
    }
    const float m = red[0]; __syncthreads();

    float ls = 0.f;
    for (int j = threadIdx.x; j < NN; j += 256) {
        float e = f1r + f2s[j];
        e = fmaxf(e, 0.01f * e);
        const float s = (arow[j] > 0) ? e : NEGINF_S;
        ls += exp2f(s - m);                          // == exp(orig - max); all-masked -> 1
    }
    red[threadIdx.x] = ls; __syncthreads();
    for (int s = 128; s > 0; s >>= 1) {
        if (threadIdx.x < s) red[threadIdx.x] += red[threadIdx.x + s];
        __syncthreads();
    }
    if (threadIdx.x == 0) { mxs[row] = m; invZ[row] = 1.0f / red[0]; }
}

// ---------------- GEMM2: out = elu(softmax(mask(e)) @ h), fused ----------------
// grid NN/16 blocks, block 128 (4 waves). Block: 16 rows x 256 cols.
// P tile (16x64 bf16) built cooperatively once per chunk into LDS (double-buffered),
// each wave consumes it via ds_load_b128 A-fragments + v_wmma against hT.
#define PLDS_STRIDE 40   // uints per P row (160B: 16B-aligned, bank-spread)

__device__ __forceinline__ unsigned int attn_pack2(int a0, float g0, int a1, float g1,
                                                   float f1r, float mr, float izr) {
    float e0 = f1r + g0; e0 = fmaxf(e0, 0.01f * e0);
    float p0 = exp2f(((a0 > 0) ? e0 : NEGINF_S) - mr) * izr;
    float e1 = f1r + g1; e1 = fmaxf(e1, 0.01f * e1);
    float p1 = exp2f(((a1 > 0) ? e1 : NEGINF_S) - mr) * izr;
    return pack_bf16x2(p0, p1);
}

__global__ __launch_bounds__(128) void gemm2_kernel(const int* __restrict__ adj,
                                                    const unsigned short* __restrict__ hT,
                                                    const float* __restrict__ f1s,
                                                    const float* __restrict__ f2s,
                                                    const float* __restrict__ mxs,
                                                    const float* __restrict__ invZ,
                                                    float* __restrict__ out) {
    __shared__ unsigned int Pbuf[2][16][PLDS_STRIDE];

    const int tid  = threadIdx.x;
    const int wave = tid >> 5;
    const int lane = tid & 31;
    const int m    = lane & 15;
    const int hh   = lane >> 4;
    const int row0 = blockIdx.x * 16;
    const int n0   = wave * 64;

    // P-build assignment: thread -> (row, 8 consecutive cols) of the 16x64 tile
    const int prow = tid >> 3;                 // 0..15
    const int pcol = (tid & 7) * 8;            // 0,8,...,56
    const float f1r = f1s[row0 + prow];
    const float mr  = mxs[row0 + prow];
    const float izr = invZ[row0 + prow];
    const int* arow = adj + (size_t)(row0 + prow) * NN;

    v8f acc[4];
    #pragma unroll
    for (int t = 0; t < 4; ++t) acc[t] = (v8f){0.f,0.f,0.f,0.f,0.f,0.f,0.f,0.f};

    int buf = 0;
    for (int j0 = 0; j0 < NN; j0 += 64, buf ^= 1) {
        // ---- phase 1: cooperative P tile build (16x64 bf16) ----
        {
            const int c = j0 + pcol;
            const int4   A0 = *(const int4*)  (arow + c);
            const int4   A1 = *(const int4*)  (arow + c + 4);
            const float4 G0 = *(const float4*)(f2s + c);
            const float4 G1 = *(const float4*)(f2s + c + 4);
            uint4 pk;
            pk.x = attn_pack2(A0.x, G0.x, A0.y, G0.y, f1r, mr, izr);
            pk.y = attn_pack2(A0.z, G0.z, A0.w, G0.w, f1r, mr, izr);
            pk.z = attn_pack2(A1.x, G1.x, A1.y, G1.y, f1r, mr, izr);
            pk.w = attn_pack2(A1.z, G1.z, A1.w, G1.w, f1r, mr, izr);
            *(uint4*)&Pbuf[buf][prow][pcol >> 1] = pk;
        }
        __syncthreads();

        // ---- phase 2: A from LDS, B from hT, 8 WMMA per wave ----
        #pragma unroll
        for (int kk = 0; kk < 64; kk += 32) {
            Frag16 a;
            const unsigned int* pr = &Pbuf[buf][m][0];
            a.q[0] = *(const uint4*)(pr + ((kk +      8 * hh) >> 1));
            a.q[1] = *(const uint4*)(pr + ((kk + 16 + 8 * hh) >> 1));
            #pragma unroll
            for (int nt = 0; nt < 4; ++nt) {
                const unsigned short* brow =
                    hT + (size_t)(n0 + nt * 16 + m) * NN + j0 + kk;
                Frag16 b;
                b.q[0] = *(const uint4*)(brow + 8 * hh);
                b.q[1] = *(const uint4*)(brow + 16 + 8 * hh);
                acc[nt] = __builtin_amdgcn_wmma_f32_16x16x32_bf16(
                    false, a.v, false, b.v, (short)0, acc[nt], false, false);
            }
        }
    }

    #pragma unroll
    for (int nt = 0; nt < 4; ++nt) {
        #pragma unroll
        for (int r = 0; r < 8; ++r) {
            const int orow = row0 + r + 8 * hh;
            const int ocol = n0 + nt * 16 + m;
            const float v = acc[nt][r];
            out[(size_t)orow * DOUT + ocol] = v > 0.f ? v : expm1f(v);  // elu
        }
    }
}

// ---------------- host ----------------
extern "C" void kernel_launch(void* const* d_in, const int* in_sizes, int n_in,
                              void* d_out, int out_size, void* d_ws, size_t ws_size,
                              hipStream_t stream) {
    (void)in_sizes; (void)n_in; (void)out_size; (void)ws_size;
    const float* x   = (const float*)d_in[0];
    const int*   adj = (const int*)  d_in[1];
    const float* W   = (const float*)d_in[2];
    const float* a1  = (const float*)d_in[3];
    const float* a2  = (const float*)d_in[4];
    float* out = (float*)d_out;

    char* ws = (char*)d_ws;
    size_t off = 0;
    auto take = [&](size_t bytes) -> char* {
        char* p = ws + off;
        off += (bytes + 255) & ~(size_t)255;
        return p;
    };
    unsigned short* xbf = (unsigned short*)take((size_t)NN * DIN * 2);     // 8 MB
    unsigned short* WT  = (unsigned short*)take((size_t)DOUT * DIN * 2);   // 256 KB
    float*          h   = (float*)         take((size_t)NN * DOUT * 4);    // 8 MB
    unsigned short* hT  = (unsigned short*)take((size_t)DOUT * NN * 2);    // 4 MB
    float* f1s = (float*)take((size_t)NN * 4);
    float* f2s = (float*)take((size_t)NN * 4);
    float* mxs = (float*)take((size_t)NN * 4);
    float* izv = (float*)take((size_t)NN * 4);

    cvt_x_kernel<<<(NN * DIN + 255) / 256, 256, 0, stream>>>(x, xbf);
    cvt_w_kernel<<<(DIN * DOUT + 255) / 256, 256, 0, stream>>>(W, WT);
    gemm1_kernel<<<dim3(NN / 16, DOUT / 32), 64, 0, stream>>>(xbf, WT, h, hT);
    f12_kernel<<<NN / 8, 256, 0, stream>>>(h, a1, a2, f1s, f2s);
    rowstats_kernel<<<NN, 256, 0, stream>>>(adj, f1s, f2s, mxs, izv);
    gemm2_kernel<<<NN / 16, 128, 0, stream>>>(adj, hT, f1s, f2s, mxs, izv, out);
}
